// SSD_14731737825578
// MI455X (gfx1250) — matrix operand order repaired and checked by reading, not compiled, not run on hardware
//
#include <hip/hip_runtime.h>

typedef __attribute__((ext_vector_type(16))) _Float16 v16h;
typedef __attribute__((ext_vector_type(8)))  float    v8f;
typedef __attribute__((ext_vector_type(2)))  int      v2i;

#define NPRIORS 7220
#define PRIORS0 5776   // 38*38*4
#define TOPK 200
#define APITCH 38      // halves; 19 dwords -> conflict-free A-frag reads
#define BPITCH 36      // halves; 18 dwords -> conflict-free B-frag reads, 8B-aligned rows

#if __has_builtin(__builtin_amdgcn_global_load_async_to_lds_b64) && \
    __has_builtin(__builtin_amdgcn_s_wait_asynccnt)
#define USE_ASYNC_LDS 1
#else
#define USE_ASYNC_LDS 0
#endif

union Frag16 { v16h h; unsigned int u[8]; };

// ---------------------------------------------------------------------------
// Per-layer weight repack: fp32 [Cout][K] -> fp16 [CoutP][KP], zero padded.
// Makes every async B-tile load unguarded, aligned and in-bounds.
// ---------------------------------------------------------------------------
__global__ void pack_weights(const float* __restrict__ w, _Float16* __restrict__ out,
                             int Cout, int K, int KP, int total) {
    int idx = blockIdx.x * 256 + threadIdx.x;
    if (idx >= total) return;
    int o = idx / KP, k = idx - o * KP;
    _Float16 v = (_Float16)0.f;
    if (o < Cout && k < K) v = (_Float16)w[(size_t)o * K + k];
    out[idx] = v;
}

// ---------------------------------------------------------------------------
// Implicit-GEMM 3x3 conv (pad=1, stride=1) using WMMA f16 -> f32.
// GEMM: M = B*S*S, N = Cout, K = 9*Cin (padded to KP).
// Block = 256 threads = 8 waves; block tile 128(M) x 128(N).
// Waves: 4 in M x 2 in N; wave tile 32x64 = 2 A-frags x 4 B-frags, 8 wmma/step.
// LDS double-buffered; A register-staged, B via async global->LDS (ASYNCcnt);
// one barrier per K-step.
// ---------------------------------------------------------------------------
template<typename OutT, bool RELU>
__global__ __launch_bounds__(256)
void conv3x3_wmma(const _Float16* __restrict__ in, OutT* __restrict__ out,
                  const _Float16* __restrict__ wgt, const float* __restrict__ bias,
                  int Cin, int Cout, int S, int M, int K, int KP)
{
    __shared__ _Float16 aT[2][128][APITCH];
    __shared__ _Float16 bT[2][128][BPITCH];

    const int tid  = threadIdx.x;
    const int lane = tid & 31;
    const int wv   = tid >> 5;
    const int wvM  = wv & 3;             // 4 waves tile M (32 rows each)
    const int wvN  = wv >> 2;            // 2 waves tile N (64 cols each)
    const int m0   = blockIdx.x * 128;
    const int n0   = blockIdx.y * 128;
    const int SS   = S * S;
    const int nsteps = KP >> 5;

    // ---- A-loader mapping: one k per thread (kidx), 16 m-rows ----
    const int kidx  = tid >> 3;          // 0..31
    const int mbase = tid & 7;
    int pixBase[16];
    unsigned int xy[16];
    #pragma unroll
    for (int i = 0; i < 16; ++i) {
        int m = m0 + mbase + 8 * i;
        if (m < M) {
            int x = m % S; int t = m / S; int y = t % S; int img = t / S;
            pixBase[i] = img * Cin * SS + y * S + x;
            xy[i] = (unsigned)x | ((unsigned)y << 16);
        } else {
            pixBase[i] = 0;
            xy[i] = 0xFFFFFFFFu;         // guarantees bounds checks fail
        }
    }

    v8f acc[2][4];
    #pragma unroll
    for (int s = 0; s < 2; ++s)
        #pragma unroll
        for (int t = 0; t < 4; ++t)
            #pragma unroll
            for (int e = 0; e < 8; ++e) acc[s][t][e] = 0.f;

    _Float16 aReg[16];

    auto stageA = [&](int k0) {
        int kk = k0 + kidx;
        int ci = kk / 9;                 // one division pair per K-step
        int r  = kk - ci * 9;
        int kh = r / 3;
        int kw = r - kh * 3;
        int off = ci * SS + (kh - 1) * S + (kw - 1);
        bool kvalid = kk < K;
        unsigned dxu = (unsigned)(kw - 1);
        unsigned dyu = (unsigned)(kh - 1);
        #pragma unroll
        for (int i = 0; i < 16; ++i) {
            unsigned x = xy[i] & 0xFFFFu;
            unsigned y = xy[i] >> 16;
            unsigned ux = x + dxu;
            unsigned uy = y + dyu;
            _Float16 v = (_Float16)0.f;
            if (kvalid && ux < (unsigned)S && uy < (unsigned)S)
                v = in[pixBase[i] + off];
            aReg[i] = v;
        }
    };
    auto commitA = [&](int buf) {
        #pragma unroll
        for (int i = 0; i < 16; ++i)
            aT[buf][mbase + 8 * i][kidx] = aReg[i];
    };

#if USE_ASYNC_LDS
    // B tile via async global->LDS: 128 rows x 32 halves; 4 x b64 per thread.
    const int brow_ = tid >> 1;
    const int bhalf = tid & 1;
    auto asyncB = [&](int k0, int buf) {
        const _Float16* gp = wgt + (size_t)(n0 + brow_) * KP + k0 + bhalf * 16;
        _Float16* lp = &bT[buf][brow_][bhalf * 16];
        __builtin_amdgcn_global_load_async_to_lds_b64(
            (__attribute__((address_space(1))) v2i*)gp,
            (__attribute__((address_space(3))) v2i*)lp, 0, 0);
        __builtin_amdgcn_global_load_async_to_lds_b64(
            (__attribute__((address_space(1))) v2i*)gp,
            (__attribute__((address_space(3))) v2i*)lp, 8, 0);
        __builtin_amdgcn_global_load_async_to_lds_b64(
            (__attribute__((address_space(1))) v2i*)gp,
            (__attribute__((address_space(3))) v2i*)lp, 16, 0);
        __builtin_amdgcn_global_load_async_to_lds_b64(
            (__attribute__((address_space(1))) v2i*)gp,
            (__attribute__((address_space(3))) v2i*)lp, 24, 0);
    };
#else
    const int oloc  = tid >> 1;
    const int khalf = tid & 1;
    _Float16 bReg[16];
    auto stageB = [&](int k0) {
        const _Float16* gp = wgt + (size_t)(n0 + oloc) * KP + k0 + khalf * 16;
        #pragma unroll
        for (int j = 0; j < 16; ++j) bReg[j] = gp[j];
    };
    auto commitB = [&](int buf) {
        #pragma unroll
        for (int j = 0; j < 8; ++j) {
            union { _Float16 h[2]; unsigned int u; } p;
            p.h[0] = bReg[2 * j]; p.h[1] = bReg[2 * j + 1];
            *(unsigned int*)&bT[buf][oloc][khalf * 16 + 2 * j] = p.u;
        }
    };
#endif

    stageA(0);
#if USE_ASYNC_LDS
    asyncB(0, 0);
#else
    stageB(0);
#endif

    for (int s = 0; s < nsteps; ++s) {
        const int buf = s & 1;
        commitA(buf);
#if USE_ASYNC_LDS
        __builtin_amdgcn_s_wait_asynccnt(0);   // own B async writes landed in LDS
#else
        commitB(buf);
#endif
        __syncthreads();
        if (s + 1 < nsteps) {                  // overlap next-tile loads w/ compute
            stageA((s + 1) * 32);
#if USE_ASYNC_LDS
            asyncB((s + 1) * 32, buf ^ 1);
#else
            stageB((s + 1) * 32);
#endif
        }

        // A fragments (16-bit A 16x32 lane layout)
        Frag16 af[2];
        #pragma unroll
        for (int sub = 0; sub < 2; ++sub) {
            const int mrow = wvM * 32 + sub * 16 + (lane & 15);
            const unsigned int* arow = (const unsigned int*)(&aT[buf][mrow][0]);
            const int kw0 = (lane >> 4) * 4;
            #pragma unroll
            for (int j = 0; j < 4; ++j) {
                af[sub].u[j]     = arow[kw0 + j];
                af[sub].u[4 + j] = arow[kw0 + 8 + j];
            }
        }
        // B fragments + 8 WMMAs
        #pragma unroll
        for (int nt = 0; nt < 4; ++nt) {
            Frag16 bf;
            const int ncol = wvN * 64 + nt * 16 + (lane & 15);
            const unsigned int* brow = (const unsigned int*)(&bT[buf][ncol][0]);
            const int kwb = (lane >> 4) * 4;
            #pragma unroll
            for (int j = 0; j < 4; ++j) {
                bf.u[j]     = brow[kwb + j];
                bf.u[4 + j] = brow[kwb + 8 + j];
            }
            acc[0][nt] = __builtin_amdgcn_wmma_f32_16x16x32_f16(
                false, af[0].h, false, bf.h, (short)0, acc[0][nt], false, false);
            acc[1][nt] = __builtin_amdgcn_wmma_f32_16x16x32_f16(
                false, af[1].h, false, bf.h, (short)0, acc[1][nt], false, false);
        }
    }

    // ---- epilogue: bias (+ReLU), scatter to NCHW; row decomposition hoisted ----
    const int colSub = lane & 15;
    const int rhalf  = (lane >> 4) << 3;     // C/D layout: M = r + 8*(lane>=16)
    #pragma unroll
    for (int sub = 0; sub < 2; ++sub) {
        const int mBase = m0 + wvM * 32 + sub * 16 + rhalf;
        int obase[8]; bool mv[8];
        #pragma unroll
        for (int r = 0; r < 8; ++r) {
            int m = mBase + r;
            mv[r] = m < M;
            int mm = mv[r] ? m : 0;
            int x = mm % S; int t2 = mm / S; int y = t2 % S; int img = t2 / S;
            obase[r] = (img * Cout) * SS + y * S + x;
        }
        #pragma unroll
        for (int nt = 0; nt < 4; ++nt) {
            int co = n0 + wvN * 64 + nt * 16 + colSub;
            if (co >= Cout) continue;
            float bi = bias[co];
            #pragma unroll
            for (int r = 0; r < 8; ++r) {
                if (!mv[r]) continue;
                float v = acc[sub][nt][r] + bi;
                if (RELU) v = fmaxf(v, 0.f);
                out[obase[r] + co * SS] = (OutT)v;
            }
        }
    }
}

// ---------------------------------------------------------------------------
__global__ void f32_to_f16(const float* __restrict__ in, _Float16* __restrict__ out, int n) {
    int i = blockIdx.x * 256 + threadIdx.x;
    if (i < n) out[i] = (_Float16)in[i];
}

__global__ void maxpool2_f16(const _Float16* __restrict__ in, _Float16* __restrict__ out,
                             int C, int Sin, int Sout, int total) {
    int i = blockIdx.x * 256 + threadIdx.x;
    if (i >= total) return;
    int x = i % Sout; int t = i / Sout; int y = t % Sout; t /= Sout;
    int c = t % C; int img = t / C;
    float m = -3.0e38f;
    #pragma unroll
    for (int dy = 0; dy < 2; ++dy)
        #pragma unroll
        for (int dx = 0; dx < 2; ++dx) {
            int iy = 2 * y + dy, ix = 2 * x + dx;
            if (iy < Sin && ix < Sin)
                m = fmaxf(m, (float)in[((img * C + c) * Sin + iy) * Sin + ix]);
        }
    out[((img * C + c) * Sout + y) * Sout + x] = (_Float16)m;
}

__global__ void decode_kernel(const float* __restrict__ loc0, const float* __restrict__ conf0,
                              const float* __restrict__ loc1, const float* __restrict__ conf1,
                              const float* __restrict__ priors,
                              float* __restrict__ boxes, float* __restrict__ scores, int B)
{
    int i = blockIdx.x * 256 + threadIdx.x;
    if (i >= B * NPRIORS) return;
    int p = i % NPRIORS; int b = i / NPRIORS;
    const float *loc, *conf; int Sd, q;
    if (p < PRIORS0) { loc = loc0; conf = conf0; Sd = 38; q = p; }
    else             { loc = loc1; conf = conf1; Sd = 19; q = p - PRIORS0; }
    int a = q & 3; int cell = q >> 2; int x = cell % Sd; int y = cell / Sd;
    float l[4];
    #pragma unroll
    for (int d = 0; d < 4; ++d)
        l[d] = loc[((b * 16 + a * 4 + d) * Sd + y) * Sd + x];
    float c0 = conf[((b * 8 + a * 2 + 0) * Sd + y) * Sd + x];
    float c1 = conf[((b * 8 + a * 2 + 1) * Sd + y) * Sd + x];
    float sc = 1.f / (1.f + __expf(c0 - c1));
    float px = priors[p * 4 + 0], py = priors[p * 4 + 1];
    float pw = priors[p * 4 + 2], ph = priors[p * 4 + 3];
    float cx = px + l[0] * 0.1f * pw;
    float cy = py + l[1] * 0.1f * ph;
    float w  = pw * __expf(l[2] * 0.2f);
    float h  = ph * __expf(l[3] * 0.2f);
    float x1 = cx - w * 0.5f, y1 = cy - h * 0.5f;
    float* bo = boxes + (size_t)(b * NPRIORS + p) * 4;
    bo[0] = x1; bo[1] = y1; bo[2] = x1 + w; bo[3] = y1 + h;
    scores[b * NPRIORS + p] = sc;
}

__device__ __forceinline__ float iou_ref(const float* A, const float* Bx) {
    float areaA = fmaxf(A[2] - A[0], 0.f) * fmaxf(A[3] - A[1], 0.f);
    float areaB = fmaxf(Bx[2] - Bx[0], 0.f) * fmaxf(Bx[3] - Bx[1], 0.f);
    float xx1 = fmaxf(A[0], Bx[0]), yy1 = fmaxf(A[1], Bx[1]);
    float xx2 = fminf(A[2], Bx[2]), yy2 = fminf(A[3], Bx[3]);
    float inter = fmaxf(xx2 - xx1, 0.f) * fmaxf(yy2 - yy1, 0.f);
    return inter / fmaxf(areaA + areaB - inter, 1e-9f);
}

__global__ __launch_bounds__(256)
void nms_kernel(const float* __restrict__ scores, const float* __restrict__ boxes,
                float* __restrict__ scratch, float* __restrict__ out)
{
    const int b = blockIdx.x;
    const int tid = threadIdx.x;
    __shared__ float sVal[256]; __shared__ int sIdx[256];
    __shared__ float topV[TOPK]; __shared__ int topI[TOPK];
    __shared__ float cand[TOPK][4]; __shared__ int keep[TOPK];

    float* sc = scratch + (size_t)b * NPRIORS;
    for (int i = tid; i < NPRIORS; i += 256) sc[i] = scores[b * NPRIORS + i];
    __syncthreads();

    for (int it = 0; it < TOPK; ++it) {
        float best = -1e30f; int bi = 0;
        for (int i = tid; i < NPRIORS; i += 256) {
            float v = sc[i];
            if (v > best) { best = v; bi = i; }
        }
        sVal[tid] = best; sIdx[tid] = bi;
        __syncthreads();
        for (int s = 128; s > 0; s >>= 1) {
            if (tid < s && sVal[tid + s] > sVal[tid]) { sVal[tid] = sVal[tid + s]; sIdx[tid] = sIdx[tid + s]; }
            __syncthreads();
        }
        if (tid == 0) { topV[it] = sVal[0]; topI[it] = sIdx[0]; sc[sIdx[0]] = -1e30f; }
        __syncthreads();
    }

    for (int i = tid; i < TOPK; i += 256) {
        const float* bx = boxes + (size_t)(b * NPRIORS + topI[i]) * 4;
        cand[i][0] = bx[0]; cand[i][1] = bx[1]; cand[i][2] = bx[2]; cand[i][3] = bx[3];
        keep[i] = topV[i] > 0.01f;
    }
    __syncthreads();
    for (int i = 0; i < TOPK; ++i) {
        if (keep[i]) {
            for (int j = tid; j < TOPK; j += 256)
                if (j > i && keep[j] && iou_ref(cand[i], cand[j]) > 0.45f) keep[j] = 0;
        }
        __syncthreads();
    }

    for (int k = tid; k < TOPK; k += 256) {
        float* o0 = out + ((size_t)(b * 2 + 0) * TOPK + k) * 5;
        #pragma unroll
        for (int d = 0; d < 5; ++d) o0[d] = 0.f;
        float* o1 = out + ((size_t)(b * 2 + 1) * TOPK + k) * 5;
        int kp = keep[k];
        o1[0] = kp ? topV[k] : 0.f;
        #pragma unroll
        for (int d = 0; d < 4; ++d) o1[1 + d] = kp ? cand[k][d] : 0.f;
    }
}

// ---------------------------------------------------------------------------
extern "C" void kernel_launch(void* const* d_in, const int* in_sizes, int n_in,
                              void* d_out, int out_size, void* d_ws, size_t ws_size,
                              hipStream_t stream)
{
    (void)in_sizes; (void)n_in; (void)out_size; (void)ws_size;
    const int B = 4;
    const float* x      = (const float*)d_in[0];
    const float* priors = (const float*)d_in[1];
    const float* vw[13]; const float* vb[13];
    for (int i = 0; i < 13; ++i) { vw[i] = (const float*)d_in[2 + 2 * i]; vb[i] = (const float*)d_in[3 + 2 * i]; }
    const float* lw[2] = { (const float*)d_in[28], (const float*)d_in[30] };
    const float* lb[2] = { (const float*)d_in[29], (const float*)d_in[31] };
    const float* cw[2] = { (const float*)d_in[32], (const float*)d_in[34] };
    const float* cb[2] = { (const float*)d_in[33], (const float*)d_in[35] };

    char* ws = (char*)d_ws;
    size_t off = 0;
    auto alloc = [&](size_t bytes) -> char* {
        char* p = ws + off; off = (off + bytes + 255) & ~(size_t)255; return p;
    };
    _Float16* bufA  = (_Float16*)alloc((size_t)B * 64 * 300 * 300 * 2);
    _Float16* bufB  = (_Float16*)alloc((size_t)B * 64 * 300 * 300 * 2);
    _Float16* feat  = (_Float16*)alloc((size_t)B * 512 * 38 * 38 * 2);
    float* loc0  = (float*)alloc((size_t)B * 16 * 38 * 38 * 4);
    float* conf0 = (float*)alloc((size_t)B * 8  * 38 * 38 * 4);
    float* loc1  = (float*)alloc((size_t)B * 16 * 19 * 19 * 4);
    float* conf1 = (float*)alloc((size_t)B * 8  * 19 * 19 * 4);
    float* boxes = (float*)alloc((size_t)B * NPRIORS * 4 * 4);
    float* scr   = (float*)alloc((size_t)B * NPRIORS * 4);
    float* nmsS  = (float*)alloc((size_t)B * NPRIORS * 4);

    // pack all weights (fp32 -> padded fp16 [CoutP][KP]) every call (deterministic)
    struct PW { _Float16* p; int KP; };
    auto packW = [&](const float* w, int Cout, int K) -> PW {
        int CoutP = (Cout + 127) & ~127;
        int KP    = (K + 31) & ~31;
        _Float16* p = (_Float16*)alloc((size_t)CoutP * KP * 2);
        int total = CoutP * KP;
        pack_weights<<<(total + 255) / 256, 256, 0, stream>>>(w, p, Cout, K, KP, total);
        return { p, KP };
    };
    const int vcin[13]  = { 3, 64, 64, 128, 128, 256, 256, 256, 512, 512, 512, 512, 512 };
    const int vcout[13] = { 64, 64, 128, 128, 256, 256, 256, 512, 512, 512, 512, 512, 512 };
    PW pvw[13];
    for (int i = 0; i < 13; ++i) pvw[i] = packW(vw[i], vcout[i], vcin[i] * 9);
    PW plw[2], pcw[2];
    for (int i = 0; i < 2; ++i) { plw[i] = packW(lw[i], 16, 512 * 9); pcw[i] = packW(cw[i], 8, 512 * 9); }

    auto conv = [&](const _Float16* in, _Float16* out, PW w, const float* bi,
                    int Cin, int Cout, int S) {
        int M = B * S * S, K = Cin * 9;
        dim3 grid((M + 127) / 128, ((Cout + 127) & ~127) / 128);
        conv3x3_wmma<_Float16, true><<<grid, 256, 0, stream>>>(in, out, w.p, bi, Cin, Cout, S, M, K, w.KP);
    };
    auto convHead = [&](const _Float16* in, float* out, PW w, const float* bi,
                        int Cin, int Cout, int S) {
        int M = B * S * S, K = Cin * 9;
        dim3 grid((M + 127) / 128, ((Cout + 127) & ~127) / 128);
        conv3x3_wmma<float, false><<<grid, 256, 0, stream>>>(in, out, w.p, bi, Cin, Cout, S, M, K, w.KP);
    };
    auto pool = [&](const _Float16* in, _Float16* out, int C, int Sin, int Sout) {
        int total = B * C * Sout * Sout;
        maxpool2_f16<<<(total + 255) / 256, 256, 0, stream>>>(in, out, C, Sin, Sout, total);
    };

    { int n = B * 3 * 300 * 300; f32_to_f16<<<(n + 255) / 256, 256, 0, stream>>>(x, bufA, n); }
    conv(bufA, bufB, pvw[0], vb[0],   3,  64, 300);
    conv(bufB, bufA, pvw[1], vb[1],  64,  64, 300);
    pool(bufA, bufB, 64, 300, 150);
    conv(bufB, bufA, pvw[2], vb[2],  64, 128, 150);
    conv(bufA, bufB, pvw[3], vb[3], 128, 128, 150);
    pool(bufB, bufA, 128, 150, 75);
    conv(bufA, bufB, pvw[4], vb[4], 128, 256, 75);
    conv(bufB, bufA, pvw[5], vb[5], 256, 256, 75);
    conv(bufA, bufB, pvw[6], vb[6], 256, 256, 75);
    pool(bufB, bufA, 256, 75, 38);                       // ceil-mode pool
    conv(bufA, bufB, pvw[7], vb[7], 256, 512, 38);
    conv(bufB, bufA, pvw[8], vb[8], 512, 512, 38);
    conv(bufA, feat, pvw[9], vb[9], 512, 512, 38);       // source 0 (conv4_3 relu)
    pool(feat, bufA, 512, 38, 19);
    conv(bufA, bufB, pvw[10], vb[10], 512, 512, 19);
    conv(bufB, bufA, pvw[11], vb[11], 512, 512, 19);
    conv(bufA, bufB, pvw[12], vb[12], 512, 512, 19);     // source 1 (conv5_3 relu)

    convHead(feat, loc0,  plw[0], lb[0], 512, 16, 38);
    convHead(feat, conf0, pcw[0], cb[0], 512,  8, 38);
    convHead(bufB, loc1,  plw[1], lb[1], 512, 16, 19);
    convHead(bufB, conf1, pcw[1], cb[1], 512,  8, 19);

    { int n = B * NPRIORS;
      decode_kernel<<<(n + 255) / 256, 256, 0, stream>>>(loc0, conf0, loc1, conf1, priors, boxes, scr, B); }
    nms_kernel<<<B, 256, 0, stream>>>(scr, boxes, nmsS, (float*)d_out);
}
